// CausalSelfAttention_37185826849441
// MI455X (gfx1250) — matrix operand order repaired and checked by reading, not compiled, not run on hardware
//
#include <hip/hip_runtime.h>
#include <hip/hip_bf16.h>

// ---------------------------------------------------------------------------
// CausalSelfAttention for MI455X (gfx1250), wave32 + v_wmma_f32_16x16x32_bf16
// All GEMM operands pre-converted to bf16; weights pre-transposed to [N][K]
// so every WMMA fragment is two contiguous 16-byte loads.
// ---------------------------------------------------------------------------

typedef __attribute__((ext_vector_type(16))) __bf16 v16bf;
typedef __attribute__((ext_vector_type(8)))  __bf16 v8bf;
typedef __attribute__((ext_vector_type(8)))  float  v8f;

#define B_   2
#define L_   2048
#define D_   2048
#define H_   16
#define KVH_ 4
#define HD_  128

// A-fragment (16x32 bf16, ISA 7.12.2): lanes 0-15 hold K {0..7,16..23},
// lanes 16-31 hold K {8..15,24..31} -> base = row + k0 + 8*half, read +0 / +16
__device__ __forceinline__ v16bf cat16(v8bf lo, v8bf hi) {
    return __builtin_shufflevector(lo, hi, 0, 1, 2, 3, 4, 5, 6, 7,
                                   8, 9, 10, 11, 12, 13, 14, 15);
}
__device__ __forceinline__ v16bf load_a_frag(const __bf16* base) {
    const v8bf* p = (const v8bf*)base;
    return cat16(p[0], p[2]);
}
// B-fragment (32x16 bf16): lanes 0-15 hold K {0..7,16..23}, lanes 16-31 K {8..15,24..31}
// -> base = col-row + k0 + 8*half, read +0 / +16
__device__ __forceinline__ v16bf load_b_frag(const __bf16* base) {
    const v8bf* p = (const v8bf*)base;
    return cat16(p[0], p[2]);
}
__device__ __forceinline__ int ka_idx(int half, int e) {
    return (e & 7) + (half << 3) + ((e >> 3) << 4);
}

// ---------------------------------------------------------------------------
// fp32 -> bf16 elementwise convert
// ---------------------------------------------------------------------------
__global__ __launch_bounds__(256) void conv_bf16_kernel(const float* __restrict__ in,
                                                        __bf16* __restrict__ out,
                                                        unsigned n) {
    const unsigned i = blockIdx.x * 256u + threadIdx.x;
    if (i < n) out[i] = (__bf16)in[i];
}

// ---------------------------------------------------------------------------
// Transpose-convert: W [K][N] fp32 -> Wt [N][K] bf16 (LDS 16x16 tile)
// ---------------------------------------------------------------------------
__global__ __launch_bounds__(256) void transconv_kernel(const float* __restrict__ W,
                                                        __bf16* __restrict__ Wt,
                                                        int K, int N) {
    __shared__ __bf16 tile[16][17];
    const int tx = (int)(threadIdx.x & 15);
    const int ty = (int)(threadIdx.x >> 4);
    const int k0 = (int)blockIdx.x * 16;
    const int n0 = (int)blockIdx.y * 16;
    tile[ty][tx] = (__bf16)W[(size_t)(k0 + ty) * N + n0 + tx];
    __syncthreads();
    Wt[(size_t)(n0 + ty) * K + k0 + tx] = tile[tx][ty];
}

// ---------------------------------------------------------------------------
// GEMM: C[M,N] = A[M,K] * Bt[N,K]^T, bf16 operands, 32x64 tile per wave
// (2 A-frags x 4 B-frags -> 8 WMMAs per 32-deep K step)
// ---------------------------------------------------------------------------
template <typename OT>
__global__ __launch_bounds__(256) void gemm_bf16t(const __bf16* __restrict__ A,
                                                  const __bf16* __restrict__ Bt,
                                                  OT* __restrict__ C,
                                                  int M, int N, int K) {
    const int wave = (int)((blockIdx.x * blockDim.x + threadIdx.x) >> 5);
    const int lane = (int)(threadIdx.x & 31);
    const int half = lane >> 4;
    const int ln   = lane & 15;
    const int tilesN = N >> 6;                 // 64-wide tiles
    const int tiles  = (M >> 5) * tilesN;      // 32-tall tiles
    if (wave >= tiles) return;
    const int tm = wave / tilesN;
    const int tn = wave - tm * tilesN;
    const int m0 = tm * 32;
    const int n0 = tn * 64;

    v8f acc[2][4];
#pragma unroll
    for (int i = 0; i < 2; ++i)
#pragma unroll
        for (int j = 0; j < 4; ++j)
#pragma unroll
            for (int v = 0; v < 8; ++v) acc[i][j][v] = 0.f;

    const __bf16* arow0 = A + (size_t)(m0 + ln) * K;
    const __bf16* arow1 = A + (size_t)(m0 + 16 + ln) * K;

    for (int k0 = 0; k0 < K; k0 += 32) {
        __builtin_prefetch((const void*)(arow0 + k0 + 128), 0, 1);
        __builtin_prefetch((const void*)(Bt + (size_t)(n0 + ln) * K + k0 + 128), 0, 1);
        v16bf a[2], b[4];
        a[0] = load_a_frag(arow0 + k0 + (half << 3));
        a[1] = load_a_frag(arow1 + k0 + (half << 3));
#pragma unroll
        for (int j = 0; j < 4; ++j)
            b[j] = load_b_frag(Bt + (size_t)(n0 + j * 16 + ln) * K + k0 + (half << 3));
#pragma unroll
        for (int i = 0; i < 2; ++i)
#pragma unroll
            for (int j = 0; j < 4; ++j)
                acc[i][j] = __builtin_amdgcn_wmma_f32_16x16x32_bf16(
                    false, a[i], false, b[j], (short)0, acc[i][j], false, false);
    }
#pragma unroll
    for (int i = 0; i < 2; ++i)
#pragma unroll
        for (int j = 0; j < 4; ++j)
#pragma unroll
            for (int v = 0; v < 8; ++v) {
                const int row = m0 + i * 16 + v + (half << 3);
                C[(size_t)row * N + n0 + j * 16 + ln] = (OT)acc[i][j][v];
            }
}

// ---------------------------------------------------------------------------
// RoPE on Q: Qraw bf16 [B*L, H*HD] -> Qbf bf16 [B,H,L,HD]
// ---------------------------------------------------------------------------
__global__ __launch_bounds__(256) void rope_q_kernel(const __bf16* __restrict__ Qraw,
                                                     __bf16* __restrict__ Qbf) {
    const unsigned idx = blockIdx.x * 256u + threadIdx.x;  // 2^22 threads
    const int d = idx & 63;
    const int l = (idx >> 6) & 2047;
    const int h = (idx >> 17) & 15;
    const int b = idx >> 21;
    float s, c;
    const float inv = __powf(10000.f, -(float)d * (1.f / 64.f));
    __sincosf((float)l * inv, &s, &c);
    const size_t src = ((size_t)(b * L_ + l)) * (H_ * HD_) + h * HD_ + d;
    const float q1 = (float)Qraw[src];
    const float q2 = (float)Qraw[src + 64];
    const size_t dst = (((size_t)(b * H_ + h)) * L_ + l) * HD_ + d;
    Qbf[dst]      = (__bf16)(q1 * c - q2 * s);
    Qbf[dst + 64] = (__bf16)(q2 * c + q1 * s);
}

// ---------------------------------------------------------------------------
// RoPE on K + relayout V: KVraw bf16 [B*L, KVH*2*HD]
//   -> Kbf bf16 [B,KVH,L,HD] (roped), Vt bf16 [B,KVH,HD,L] (transposed)
// ---------------------------------------------------------------------------
__global__ __launch_bounds__(256) void rope_kv_kernel(const __bf16* __restrict__ KVraw,
                                                      __bf16* __restrict__ Kbf,
                                                      __bf16* __restrict__ Vt) {
    const unsigned idx = blockIdx.x * 256u + threadIdx.x;  // 2^20 threads
    const int d   = idx & 63;
    const int l   = (idx >> 6) & 2047;
    const int kvh = (idx >> 17) & 3;
    const int b   = idx >> 19;
    float s, c;
    const float inv = __powf(10000.f, -(float)d * (1.f / 64.f));
    __sincosf((float)l * inv, &s, &c);
    const size_t src = ((size_t)(b * L_ + l)) * (KVH_ * 2 * HD_) + kvh * (2 * HD_);
    const float k1 = (float)KVraw[src + d];
    const float k2 = (float)KVraw[src + d + 64];
    const float v1 = (float)KVraw[src + HD_ + d];
    const float v2 = (float)KVraw[src + HD_ + d + 64];
    const size_t kdst = (((size_t)(b * KVH_ + kvh)) * L_ + l) * HD_ + d;
    Kbf[kdst]      = (__bf16)(k1 * c - k2 * s);
    Kbf[kdst + 64] = (__bf16)(k2 * c + k1 * s);
    const size_t vbase = ((size_t)(b * KVH_ + kvh)) * HD_;
    Vt[(vbase + d) * L_ + l]      = (__bf16)v1;
    Vt[(vbase + d + 64) * L_ + l] = (__bf16)v2;
}

// ---------------------------------------------------------------------------
// Flash attention: one wave per (b, h, 16-query tile); online softmax over
// 32-key blocks; QK^T and PV via bf16 WMMA; P transposed through private LDS.
// V consumed from transposed layout so PV B-fragments are contiguous.
// ---------------------------------------------------------------------------
__global__ __launch_bounds__(256) void flash_attn_kernel(const __bf16* __restrict__ Q,
                                                         const __bf16* __restrict__ Kb,
                                                         const __bf16* __restrict__ Vt,
                                                         const int* __restrict__ amask,
                                                         __bf16* __restrict__ Y) {
    __shared__ __bf16 plds[8][16 * 32];   // per-wave P tile (16 q x 32 keys)
    const int wv   = (int)(threadIdx.x >> 5);
    const int lane = (int)(threadIdx.x & 31);
    const int half = lane >> 4;
    const int ln   = lane & 15;

    const int gidx = (int)blockIdx.x * 8 + wv;       // 4096 waves total
    const int qt = gidx & 127;                       // L/16
    const int h  = (gidx >> 7) & 15;
    const int b  = gidx >> 11;
    const int q0 = qt * 16;
    const int kvh = h >> 2;                          // GQA: 4 q-heads per kv-head

    const __bf16* Qh  = Q  + (((size_t)(b * H_  + h  )) * L_) * HD_;
    const __bf16* Kh  = Kb + (((size_t)(b * KVH_ + kvh)) * L_) * HD_;
    const __bf16* Vth = Vt + (((size_t)(b * KVH_ + kvh)) * HD_) * L_;

    // Q fragments (16 rows x 128 dims = 4 chunks of K=32)
    v16bf qf[4];
#pragma unroll
    for (int c = 0; c < 4; ++c)
        qf[c] = load_a_frag(Qh + (size_t)(q0 + ln) * HD_ + c * 32 + (half << 3));

    float mrow[8], lrow[8];
    v8f o[8];
#pragma unroll
    for (int v = 0; v < 8; ++v) { mrow[v] = -1e30f; lrow[v] = 0.f; }
#pragma unroll
    for (int t = 0; t < 8; ++t)
#pragma unroll
        for (int v = 0; v < 8; ++v) o[t][v] = 0.f;

    const float scale = 0.08838834764831845f;        // 1/sqrt(128)
    const int nblk = (q0 + 16 + 31) >> 5;            // causal: keys <= q0+15

    for (int blk = 0; blk < nblk; ++blk) {
        const int ks = blk << 5;
        v8f s[2];
#pragma unroll
        for (int t = 0; t < 2; ++t) {
            v8f st;
#pragma unroll
            for (int v = 0; v < 8; ++v) st[v] = 0.f;
            const int key = ks + t * 16 + ln;        // B-fragment column
#pragma unroll
            for (int c = 0; c < 4; ++c) {
                const v16bf kf = load_b_frag(Kh + (size_t)key * HD_ + c * 32 + (half << 3));
                st = __builtin_amdgcn_wmma_f32_16x16x32_bf16(false, qf[c], false, kf,
                                                             (short)0, st, false, false);
            }
            const int km = amask[b * L_ + key];
#pragma unroll
            for (int v = 0; v < 8; ++v) {
                const int q = q0 + v + (half << 3);
                float val = st[v] * scale;
                if (key > q || km == 0) val = -1e30f;
                st[v] = val;
            }
            s[t] = st;
        }
        // online softmax row statistics (rows live in 16-lane halves)
        float alpha[8];
#pragma unroll
        for (int v = 0; v < 8; ++v) {
            float mx = fmaxf(s[0][v], s[1][v]);
            mx = fmaxf(mx, __shfl_xor(mx, 1));
            mx = fmaxf(mx, __shfl_xor(mx, 2));
            mx = fmaxf(mx, __shfl_xor(mx, 4));
            mx = fmaxf(mx, __shfl_xor(mx, 8));
            const float mnew = fmaxf(mrow[v], mx);
            alpha[v] = expf(mrow[v] - mnew);
            mrow[v]  = mnew;
            const float p0 = expf(s[0][v] - mnew);
            const float p1 = expf(s[1][v] - mnew);
            s[0][v] = p0; s[1][v] = p1;
            float rs = p0 + p1;
            rs += __shfl_xor(rs, 1);
            rs += __shfl_xor(rs, 2);
            rs += __shfl_xor(rs, 4);
            rs += __shfl_xor(rs, 8);
            lrow[v] = lrow[v] * alpha[v] + rs;
        }
#pragma unroll
        for (int t = 0; t < 8; ++t)
#pragma unroll
            for (int v = 0; v < 8; ++v) o[t][v] *= alpha[v];

        // P tile: C-layout -> LDS -> A-layout (transpose through LDS)
#pragma unroll
        for (int t = 0; t < 2; ++t)
#pragma unroll
            for (int v = 0; v < 8; ++v)
                plds[wv][(v + (half << 3)) * 32 + t * 16 + ln] = (__bf16)s[t][v];
        const v16bf pf = load_a_frag(&plds[wv][ln * 32 + (half << 3)]);

        // O += P(16x32) * V(32x128), 8 column tiles; Vt rows are contiguous keys
#pragma unroll
        for (int nt = 0; nt < 8; ++nt) {
            const v16bf vf = load_b_frag(Vth + (size_t)(nt * 16 + ln) * L_ + ks + (half << 3));
            o[nt] = __builtin_amdgcn_wmma_f32_16x16x32_bf16(false, pf, false, vf,
                                                            (short)0, o[nt], false, false);
        }
    }

    // normalize + write Y [B, L, H*HD] bf16
#pragma unroll
    for (int nt = 0; nt < 8; ++nt)
#pragma unroll
        for (int v = 0; v < 8; ++v) {
            const int q = q0 + v + (half << 3);
            const float den = lrow[v];
            const float val = (den > 0.f) ? o[nt][v] / den : 0.f;
            Y[((size_t)(b * L_ + q)) * (H_ * HD_) + h * HD_ + nt * 16 + ln] = (__bf16)val;
        }
}

// ---------------------------------------------------------------------------
// Host launch
// ---------------------------------------------------------------------------
extern "C" void kernel_launch(void* const* d_in, const int* in_sizes, int n_in,
                              void* d_out, int out_size, void* d_ws, size_t ws_size,
                              hipStream_t stream) {
    (void)in_sizes; (void)n_in; (void)out_size; (void)ws_size;
    const float* x   = (const float*)d_in[0];
    const int*   am  = (const int*)d_in[1];
    const float* Wq  = (const float*)d_in[2];
    const float* Wkv = (const float*)d_in[3];
    const float* Wo  = (const float*)d_in[4];
    float* out = (float*)d_out;

    const size_t MB = (size_t)1 << 20;
    char* w = (char*)d_ws;
    __bf16* xbf   = (__bf16*)(w);             // [4096,2048]       16 MB
    __bf16* Wqt   = (__bf16*)(w + 16 * MB);   // [2048,2048]        8 MB
    __bf16* Wkvt  = (__bf16*)(w + 24 * MB);   // [1024,2048]        4 MB
    __bf16* Wot   = (__bf16*)(w + 28 * MB);   // [2048,2048]        8 MB
    __bf16* Qraw  = (__bf16*)(w + 36 * MB);   // [4096,2048]       16 MB
    __bf16* KVraw = (__bf16*)(w + 52 * MB);   // [4096,1024]        8 MB
    __bf16* Qbf   = (__bf16*)(w + 60 * MB);   // [B,H,L,HD]        16 MB
    __bf16* Kbf   = (__bf16*)(w + 76 * MB);   // [B,KVH,L,HD]       4 MB
    __bf16* Vt    = (__bf16*)(w + 80 * MB);   // [B,KVH,HD,L]       4 MB
    __bf16* Ybf   = (__bf16*)(w + 84 * MB);   // [B,L,H*HD]        16 MB

    const int M = B_ * L_;   // 4096

    // One-time operand preparation (bf16 convert + weight transpose)
    conv_bf16_kernel<<<32768, 256, 0, stream>>>(x, xbf, (unsigned)(M * D_));
    transconv_kernel<<<dim3(D_ / 16, (H_ * HD_) / 16), 256, 0, stream>>>(Wq, Wqt, D_, H_ * HD_);
    transconv_kernel<<<dim3(D_ / 16, (KVH_ * 2 * HD_) / 16), 256, 0, stream>>>(Wkv, Wkvt, D_, KVH_ * 2 * HD_);
    transconv_kernel<<<dim3((H_ * HD_) / 16, D_ / 16), 256, 0, stream>>>(Wo, Wot, H_ * HD_, D_);

    // Projections (32x64 tile per wave): waves = (M/32)*(N/64), blocks = waves/8
    gemm_bf16t<__bf16><<<512, 256, 0, stream>>>(xbf, Wqt, Qraw, M, H_ * HD_, D_);
    gemm_bf16t<__bf16><<<256, 256, 0, stream>>>(xbf, Wkvt, KVraw, M, KVH_ * 2 * HD_, D_);

    // RoPE + relayout
    rope_q_kernel<<<16384, 256, 0, stream>>>(Qraw, Qbf);
    rope_kv_kernel<<<4096, 256, 0, stream>>>(KVraw, Kbf, Vt);

    // Flash attention (4096 waves = 512 blocks of 8)
    flash_attn_kernel<<<512, 256, 0, stream>>>(Qbf, Kbf, Vt, am, Ybf);

    // Output projection
    gemm_bf16t<float><<<512, 256, 0, stream>>>(Ybf, Wot, out, M, D_, H_ * HD_);
}
